// CrossAttention_DenseAVInteractions_20538533609564
// MI455X (gfx1250) — compile-verified, matrix operand order checked by reading, not run on hardware
//
#include <hip/hip_runtime.h>

// ---------------------------------------------------------------------------
// CDNA5 (gfx1250) implementation of CrossAttention_DenseAVInteractions.
//
// Math identity exploited (keys/values are outer-sums over (i,j) pairs):
//   logits[q,(i,j)] = Sv[q,i] + Sa[q,j]   with Sv = Q@Kv^T, Sa = Q@Ka^T
//   => softmax factorizes: attn[q,i,j] = Pv[q,i] * Pa[q,j]
//   => out_head       = Pv @ Vv + Pa @ Va
// All GEMMs run through v_wmma_f32_16x16x32_f16 (fp32 accumulate).
// ---------------------------------------------------------------------------

typedef __attribute__((ext_vector_type(16))) _Float16 v16h;
typedef __attribute__((ext_vector_type(8)))  _Float16 h8;
typedef __attribute__((ext_vector_type(8)))  float    v8f;

#define TBM 64
#define TBN 64
#define TBK 32
#define LDSP 40   // 32 + 8 halfword pad (80B row) -> conflict-free ds_load_b128

// Build a 16-element f16 A/B fragment from LDS.
// ISA 7.12.2 (16-bit A 16x32): lane half h: regs 0-3 hold K = 8h..8h+7,
// regs 4-7 hold K = 16+8h..16+8h+7  => two contiguous 16B LDS reads.
__device__ __forceinline__ v16h frag_ld(const _Float16* p) {
  h8 lo = *(const h8*)(p);
  h8 hi = *(const h8*)(p + 16);
  v16h f;
#pragma unroll
  for (int e = 0; e < 8; ++e) { f[e] = lo[e]; f[e + 8] = hi[e]; }
  return f;
}

// Generic batched GEMM: C[z] = (acc? C[z] : 0) + A[z] @ op(B[z]) (+ bias)
// A: [M,K] fp32 (lda), B: [K,N] fp32 (ldb) or [N,K] if transB, C: [M,N] fp32.
// Batch z split as zo = z / bdiv, zi = z % bdiv with separate strides, so
// per-(batch,head) sub-matrices of packed tensors can be addressed directly.
__global__ __launch_bounds__(128)
void gemm_wmma_f16(const float* __restrict__ Abase,
                   const float* __restrict__ Bbase,
                   float* __restrict__ Cbase,
                   const float* __restrict__ bias,
                   int M, int N, int K,
                   int lda, int ldb, int ldc,
                   int bdiv,
                   long long sAo, long long sAi,
                   long long sBo, long long sBi,
                   long long sCo, long long sCi,
                   int transB, int accumulate) {
  const int z  = blockIdx.z;
  const int zo = z / bdiv, zi = z % bdiv;
  const float* A  = Abase + zo * sAo + zi * sAi;
  const float* Bm = Bbase + zo * sBo + zi * sBi;
  float*       C  = Cbase + zo * sCo + zi * sCi;

  const int tid  = threadIdx.x;
  const int wave = tid >> 5, lane = tid & 31;
  const int half = lane >> 4, ml = lane & 15;
  const int row0 = blockIdx.y * TBM, col0 = blockIdx.x * TBN;

  __shared__ __align__(16) _Float16 Ash[TBM * LDSP];          // [m][k]
  __shared__ __align__(16) _Float16 Bsh[TBN * LDSP];          // [n][k]

  v8f acc[4];
#pragma unroll
  for (int nt = 0; nt < 4; ++nt)
#pragma unroll
    for (int r = 0; r < 8; ++r) acc[nt][r] = 0.0f;

  for (int k0 = 0; k0 < K; k0 += TBK) {
    __syncthreads();
    const bool fullK = (k0 + TBK) <= K;

    // ---- stage A tile (64 x 32 fp32 -> f16), 512 float4 slots, 4/thread ----
#pragma unroll
    for (int i = 0; i < 4; ++i) {
      int idx = tid + i * 128;
      int r = idx >> 3, c4 = (idx & 7) << 2;
      int row = row0 + r;
      float4 v = make_float4(0.f, 0.f, 0.f, 0.f);
      if (row < M) {
        const float* src = A + (long long)row * lda + k0 + c4;
        if (fullK) {
          v = *(const float4*)src;
        } else {
          if (k0 + c4 + 0 < K) v.x = src[0];
          if (k0 + c4 + 1 < K) v.y = src[1];
          if (k0 + c4 + 2 < K) v.z = src[2];
          if (k0 + c4 + 3 < K) v.w = src[3];
        }
      }
      _Float16* d = &Ash[r * LDSP + c4];
      d[0] = (_Float16)v.x; d[1] = (_Float16)v.y;
      d[2] = (_Float16)v.z; d[3] = (_Float16)v.w;
    }

    // ---- stage B tile into Bsh[n][k] ----
    if (!transB) {
      // B row-major [K,N]: coalesced along n, scatter-transpose into LDS.
#pragma unroll
      for (int i = 0; i < 4; ++i) {
        int idx = tid + i * 128;
        int kk = idx >> 4, n4 = (idx & 15) << 2;
        int k = k0 + kk;
        float4 v = make_float4(0.f, 0.f, 0.f, 0.f);
        if (k < K) {
          const float* src = Bm + (long long)k * ldb + col0 + n4;
          if (col0 + n4 + 3 < N) {
            v = *(const float4*)src;
          } else {
            if (col0 + n4 + 0 < N) v.x = src[0];
            if (col0 + n4 + 1 < N) v.y = src[1];
            if (col0 + n4 + 2 < N) v.z = src[2];
            if (col0 + n4 + 3 < N) v.w = src[3];
          }
        }
        Bsh[(n4 + 0) * LDSP + kk] = (_Float16)v.x;
        Bsh[(n4 + 1) * LDSP + kk] = (_Float16)v.y;
        Bsh[(n4 + 2) * LDSP + kk] = (_Float16)v.z;
        Bsh[(n4 + 3) * LDSP + kk] = (_Float16)v.w;
      }
    } else {
      // B row-major [N,K] (i.e. C = A @ B^T): rows are already [n][k].
#pragma unroll
      for (int i = 0; i < 4; ++i) {
        int idx = tid + i * 128;
        int n = idx >> 3, c4 = (idx & 7) << 2;
        int row = col0 + n;
        float4 v = make_float4(0.f, 0.f, 0.f, 0.f);
        if (row < N) {
          const float* src = Bm + (long long)row * ldb + k0 + c4;
          if (fullK) {
            v = *(const float4*)src;
          } else {
            if (k0 + c4 + 0 < K) v.x = src[0];
            if (k0 + c4 + 1 < K) v.y = src[1];
            if (k0 + c4 + 2 < K) v.z = src[2];
            if (k0 + c4 + 3 < K) v.w = src[3];
          }
        }
        _Float16* d = &Bsh[n * LDSP + c4];
        d[0] = (_Float16)v.x; d[1] = (_Float16)v.y;
        d[2] = (_Float16)v.z; d[3] = (_Float16)v.w;
      }
    }
    __syncthreads();

    // ---- 4 WMMAs: this wave's 16-row strip x four 16-col subtiles ----
    v16h af = frag_ld(&Ash[(wave * 16 + ml) * LDSP + half * 8]);
#pragma unroll
    for (int nt = 0; nt < 4; ++nt) {
      v16h bf = frag_ld(&Bsh[(nt * 16 + ml) * LDSP + half * 8]);
      acc[nt] = __builtin_amdgcn_wmma_f32_16x16x32_f16(
          false, af, false, bf, (short)0, acc[nt], false, false);
    }
  }

  // ---- epilogue: C layout (ISA 7.12.2): reg r, lane l -> M=r+8*(l>>4), N=l&15
#pragma unroll
  for (int nt = 0; nt < 4; ++nt) {
    int n = col0 + nt * 16 + ml;
    if (n >= N) continue;
#pragma unroll
    for (int r = 0; r < 8; ++r) {
      int m = row0 + wave * 16 + r + half * 8;
      if (m >= M) continue;
      long long off = (long long)m * ldc + n;
      float v = acc[nt][r];
      if (accumulate) v += C[off];
      if (bias) v += bias[n];
      C[off] = v;
    }
  }
}

// In-place row softmax of softmax(scale * x) for rows of width W <= 256.
__global__ __launch_bounds__(256)
void softmax_rows(float* __restrict__ S, int W, float scale) {
  __shared__ float red[256];
  float* p = S + (long long)blockIdx.x * W;
  const int t = threadIdx.x;
  float x = (t < W) ? p[t] : -3.0e38f;
  red[t] = x; __syncthreads();
#pragma unroll
  for (int s = 128; s > 0; s >>= 1) {
    if (t < s) red[t] = fmaxf(red[t], red[t + s]);
    __syncthreads();
  }
  const float mx = red[0];
  __syncthreads();
  float e = (t < W) ? __expf(scale * (x - mx)) : 0.0f;
  red[t] = e; __syncthreads();
#pragma unroll
  for (int s = 128; s > 0; s >>= 1) {
    if (t < s) red[t] += red[t + s];
    __syncthreads();
  }
  const float inv = 1.0f / red[0];
  if (t < W) p[t] = e * inv;
}

// attn[bhq, i*64 + j] = Pv[bhq, i] * Pa[bhq, j]  -- pure streaming f4 stores.
__global__ __launch_bounds__(256)
void attn_outer(const float* __restrict__ Pv, const float* __restrict__ Pa,
                float* __restrict__ attn, long long total4) {
  long long idx = (long long)blockIdx.x * 256 + threadIdx.x;
  if (idx >= total4) return;
  long long bhq = idx / 3136;              // 12544/4 float4 per (b,h,q) row
  int r  = (int)(idx - bhq * 3136);
  int i  = r >> 4;                         // 16 float4 (=64 j) per i
  int j0 = (r & 15) << 2;
  float pv = Pv[bhq * 196 + i];
  float4 pa = *(const float4*)(Pa + bhq * 64 + j0);
  float4 o;
  o.x = pv * pa.x; o.y = pv * pa.y; o.z = pv * pa.z; o.w = pv * pa.w;
  *((float4*)attn + idx) = o;
}

extern "C" void kernel_launch(void* const* d_in, const int* in_sizes, int n_in,
                              void* d_out, int out_size, void* d_ws, size_t ws_size,
                              hipStream_t stream) {
  (void)in_sizes; (void)n_in; (void)out_size; (void)ws_size;
  const float* xmm   = (const float*)d_in[0];   // [4,196,768]
  const float* xa    = (const float*)d_in[1];   // [4, 64,768]
  const float* xv    = (const float*)d_in[2];   // [4,196,768]
  const float* Wq    = (const float*)d_in[3];   // [768,768]
  const float* Wkv   = (const float*)d_in[4];   // [1536,1536]
  const float* Wproj = (const float*)d_in[5];   // [768,768]
  const float* bproj = (const float*)d_in[6];   // [768]
  float* out = (float*)d_out;                   // out[4,196,768] ++ attn[4,12,196,12544]

  const float scale = 0.125f;                   // (768/12)^-0.5

  // workspace carve-out (fp32)
  float* Q   = (float*)d_ws;                    // [784, 768]
  float* KVv = Q   + 784LL * 768;               // [784,1536]
  float* KVa = KVv + 784LL * 1536;              // [256,1536]
  float* Sv  = KVa + 256LL * 1536;              // [48,196,196] -> Pv
  float* Sa  = Sv  + 48LL * 196 * 196;          // [48,196, 64] -> Pa
  float* O   = Sa  + 48LL * 196 * 64;           // [784, 768] (pre-proj, head-interleaved)

  const dim3 blk(128);

  // 1) Q = xmm @ Wq                      [784,768]
  gemm_wmma_f16<<<dim3(12, 13, 1), blk, 0, stream>>>(
      xmm, Wq, Q, nullptr, 784, 768, 768, 768, 768, 768,
      1, 0, 0, 0, 0, 0, 0, /*transB=*/0, /*acc=*/0);

  // 2) KVv = xv @ Wkv[:768,:]            [784,1536]
  gemm_wmma_f16<<<dim3(24, 13, 1), blk, 0, stream>>>(
      xv, Wkv, KVv, nullptr, 784, 1536, 768, 768, 1536, 1536,
      1, 0, 0, 0, 0, 0, 0, 0, 0);

  // 3) KVa = xa @ Wkv[768:,:]            [256,1536]
  gemm_wmma_f16<<<dim3(24, 4, 1), blk, 0, stream>>>(
      xa, Wkv + 768LL * 1536, KVa, nullptr, 256, 1536, 768, 768, 1536, 1536,
      1, 0, 0, 0, 0, 0, 0, 0, 0);

  // 4) Sv[z] = Q_h @ Kv_h^T              z = b*12+h, [196,196], K=64
  gemm_wmma_f16<<<dim3(4, 4, 48), blk, 0, stream>>>(
      Q, KVv, Sv, nullptr, 196, 196, 64, 768, 1536, 196,
      12, 196LL * 768, 64, 196LL * 1536, 64,
      12LL * 196 * 196, 196LL * 196, /*transB=*/1, 0);

  // 5) Sa[z] = Q_h @ Ka_h^T              [196,64], K=64
  gemm_wmma_f16<<<dim3(1, 4, 48), blk, 0, stream>>>(
      Q, KVa, Sa, nullptr, 196, 64, 64, 768, 1536, 64,
      12, 196LL * 768, 64, 64LL * 1536, 64,
      12LL * 196 * 64, 196LL * 64, /*transB=*/1, 0);

  // 6) factorized softmaxes (in place): Pv over 196, Pa over 64
  softmax_rows<<<dim3(48 * 196), dim3(256), 0, stream>>>(Sv, 196, scale);
  softmax_rows<<<dim3(48 * 196), dim3(256), 0, stream>>>(Sa, 64, scale);

  // 7) O_h = Pv @ Vv_h                   (Vv = KVv cols 768+h*64), K=196
  gemm_wmma_f16<<<dim3(1, 4, 48), blk, 0, stream>>>(
      Sv, KVv + 768, O, nullptr, 196, 64, 196, 196, 1536, 768,
      12, 12LL * 196 * 196, 196LL * 196, 196LL * 1536, 64,
      196LL * 768, 64, 0, /*acc=*/0);

  // 8) O_h += Pa @ Va_h                  (Va = KVa cols 768+h*64), K=64
  gemm_wmma_f16<<<dim3(1, 4, 48), blk, 0, stream>>>(
      Sa, KVa + 768, O, nullptr, 196, 64, 64, 64, 1536, 768,
      12, 12LL * 196 * 64, 196LL * 64, 64LL * 1536, 64,
      196LL * 768, 64, 0, /*acc=*/1);

  // 9) attn = Pv (outer) Pa -> d_out tail (472 MB streaming float4 stores)
  const long long total4 = 48LL * 196 * 12544 / 4;   // 29,503,488
  attn_outer<<<dim3((unsigned)((total4 + 255) / 256)), dim3(256), 0, stream>>>(
      Sv, Sa, out + 4LL * 196 * 768, total4);

  // 10) out = O @ Wproj + bproj          [784,768]
  gemm_wmma_f16<<<dim3(12, 13, 1), blk, 0, stream>>>(
      O, Wproj, out, bproj, 784, 768, 768, 768, 768, 768,
      1, 0, 0, 0, 0, 0, 0, 0, 0);
}
